// AutoEncoder_26096221291030
// MI455X (gfx1250) — compile-verified, matrix operand order checked by reading, not run on hardware
//
#include <hip/hip_runtime.h>
#include <hip/hip_bf16.h>

typedef __bf16 bf16_t;
typedef __attribute__((ext_vector_type(16))) __bf16 v16bf;
typedef __attribute__((ext_vector_type(8)))  __bf16 v8bf;
typedef __attribute__((ext_vector_type(8)))  float  v8f;

#define T_STEPS 64
#define BATCH   64
#define EDIM    256
#define HDIM    512
#define G3      1536   // 3*H
#define VOCAB   32000
#define CATD    768    // H + E

// ---- WMMA helpers -----------------------------------------------------------
// A-fragment (16x32 bf16) per ISA 7.12.2: lane r=lane&15 holds row r.
// elems 0..7  -> K = k0 + (lane>=16 ? 8 : 0) + i
// elems 8..15 -> K = k0 + 16 + (lane>=16 ? 8 : 0) + i
// So from a row pointer already offset by (k0 + half*8): read p[0..7], p[16..23].
static __device__ __forceinline__ v16bf frag_from_bf16row(const bf16_t* p) {
  union { v16bf v; v8bf h[2]; } u;
  u.h[0] = *reinterpret_cast<const v8bf*>(p);
  u.h[1] = *reinterpret_cast<const v8bf*>(p + 16);
  return u.v;
}

static __device__ __forceinline__ v16bf frag_from_f32row(const float* p) {
  v16bf v;
#pragma unroll
  for (int i = 0; i < 8; ++i) v[i] = (bf16_t)p[i];
#pragma unroll
  for (int i = 0; i < 8; ++i) v[8 + i] = (bf16_t)p[16 + i];
  return v;
}

static __device__ __forceinline__ v8f wmma_bf16(v16bf a, v16bf b, v8f c) {
  // (neg_a, A, neg_b, B, c_mod, C, reuse_a, reuse_b)
  return __builtin_amdgcn_wmma_f32_16x16x32_bf16(false, a, false, b, (short)0, c,
                                                 false, false);
}

// ---- CDNA5 async global->LDS DMA (ASYNCcnt path, ISA ch.10 / 15.18) --------
// Per-lane 16B transfer, DMA writes LDS directly (no VGPR round-trip).
// Generic shared pointers carry the wave-relative LDS offset in their low 32
// bits (shared aperture occupies the high 32), which is what VDST wants.
static __device__ __forceinline__ void async_copy16(bf16_t* lds_dst,
                                                    const bf16_t* gsrc) {
  asm volatile("global_load_async_to_lds_b128 %0, %1, off"
               :
               : "v"((unsigned)(size_t)lds_dst),
                 "v"((unsigned long long)(size_t)gsrc)
               : "memory");
}
static __device__ __forceinline__ void async_wait0() {
  asm volatile("s_wait_asynccnt 0" ::: "memory");
}

// ---- Kernel 1: embedding gather + input-gate GEMM (enc + dec fused) --------
// Rows 0..4095: xg[t*B+b] = emb[x[t,b]] @ encWih^T + encBih
// Rows 4096..8127: yg[r]  = emb[y[r]]  @ decWih^T + decBih  (r = t*B+b, t<63)
// Dec rows also drop bf16(emb) into catbuf cols [H..H+E).
__global__ void gates_kernel(const int* __restrict__ x, const int* __restrict__ y,
                             const float* __restrict__ emb,
                             const float* __restrict__ encWih, const float* __restrict__ encBih,
                             const float* __restrict__ decWih, const float* __restrict__ decBih,
                             float* __restrict__ xg, float* __restrict__ yg,
                             bf16_t* __restrict__ catbuf) {
  __shared__ alignas(16) bf16_t At[16][264];   // 16 rows x 256 K (+8 pad)
  const int m0  = blockIdx.y * 16;
  const bool dec = (m0 >= T_STEPS * BATCH);
  const int tid = threadIdx.x;

  {  // stage 16 embedding rows into LDS as bf16 (gather + convert => VALU path)
    const int row = tid >> 4;
    const int c0  = (tid & 15) * 16;
    const int m   = m0 + row;
    const int tok = dec ? y[m - T_STEPS * BATCH] : x[m];
    const float* er = emb + (size_t)tok * EDIM + c0;
#pragma unroll
    for (int i = 0; i < 16; ++i) {
      bf16_t bv = (bf16_t)er[i];
      At[row][c0 + i] = bv;
      if (dec)
        catbuf[(size_t)(m - T_STEPS * BATCH) * CATD + HDIM + c0 + i] = bv;
    }
  }
  __syncthreads();

  const int wave = tid >> 5, lane = tid & 31;
  const int r = lane & 15, half = lane >> 4;
  const int n = blockIdx.x * 128 + wave * 16 + r;
  const float* Wih = dec ? decWih : encWih;
  const float* bih = dec ? decBih : encBih;

  v8f acc = {0.f, 0.f, 0.f, 0.f, 0.f, 0.f, 0.f, 0.f};
#pragma unroll
  for (int k0 = 0; k0 < EDIM; k0 += 32) {
    v16bf a = frag_from_bf16row(&At[r][k0 + half * 8]);
    v16bf b = frag_from_f32row(&Wih[(size_t)n * EDIM + k0 + half * 8]);
    acc = wmma_bf16(a, b, acc);
  }
  const float bv = bih[n];
  float* out = dec ? yg : xg;
  const int rowbase = dec ? (m0 - T_STEPS * BATCH) : m0;
#pragma unroll
  for (int vv = 0; vv < 8; ++vv)
    out[(size_t)(rowbase + vv + half * 8) * G3 + n] = acc[vv] + bv;
}

// ---- Kernel 2: GRU hidden GEMM: hg[64,3H] = h[64,H] @ Whh^T + bhh ----------
__global__ void gru_matmul_kernel(const float* __restrict__ h,
                                  const float* __restrict__ Whh,
                                  const float* __restrict__ bhh,
                                  float* __restrict__ hg) {
  __shared__ alignas(16) bf16_t At[16][520];   // 16 rows x 512 K (+8 pad)
  const int m0 = blockIdx.y * 16;
  const int tid = threadIdx.x;
  {
    const int row = tid >> 4;
    const int c0  = (tid & 15) * 32;
    const float* hr = h + (size_t)(m0 + row) * HDIM + c0;
#pragma unroll
    for (int i = 0; i < 32; ++i) At[row][c0 + i] = (bf16_t)hr[i];
  }
  __syncthreads();

  const int wave = tid >> 5, lane = tid & 31;
  const int r = lane & 15, half = lane >> 4;
  const int n = blockIdx.x * 128 + wave * 16 + r;

  v8f acc = {0.f, 0.f, 0.f, 0.f, 0.f, 0.f, 0.f, 0.f};
#pragma unroll
  for (int k0 = 0; k0 < HDIM; k0 += 32) {
    v16bf a = frag_from_bf16row(&At[r][k0 + half * 8]);
    v16bf b = frag_from_f32row(&Whh[(size_t)n * HDIM + k0 + half * 8]);
    acc = wmma_bf16(a, b, acc);
  }
  const float bv = bhh[n];
#pragma unroll
  for (int vv = 0; vv < 8; ++vv)
    hg[(size_t)(m0 + vv + half * 8) * G3 + n] = acc[vv] + bv;
}

// ---- Kernel 3: GRU pointwise gate math (f32) -------------------------------
__global__ void gru_pointwise_kernel(const float* __restrict__ xg_t,
                                     const float* __restrict__ hg,
                                     float* __restrict__ h,
                                     const int* __restrict__ lens, int t,
                                     bf16_t* __restrict__ cat_t) {
  const int idx = blockIdx.x * blockDim.x + threadIdx.x;  // 0..32767
  const int b = idx >> 9, j = idx & 511;
  const float* xr = xg_t + (size_t)b * G3;
  const float* hr = hg + (size_t)b * G3;
  const float rg = 1.f / (1.f + expf(-(xr[j] + hr[j])));
  const float zg = 1.f / (1.f + expf(-(xr[HDIM + j] + hr[HDIM + j])));
  const float ng = tanhf(xr[2 * HDIM + j] + rg * hr[2 * HDIM + j]);
  const float hold = h[idx];
  float hnew = (1.f - zg) * ng + zg * hold;
  if (lens && t >= lens[b]) hnew = hold;   // packed-seq freeze (encoder)
  h[idx] = hnew;
  if (cat_t) cat_t[(size_t)b * CATD + j] = (bf16_t)hnew;  // decoder hidden out
}

// ---- Utility kernels -------------------------------------------------------
__global__ void f32_to_bf16_kernel(const float* __restrict__ src,
                                   bf16_t* __restrict__ dst, size_t n) {
  size_t i = (size_t)blockIdx.x * blockDim.x + threadIdx.x;
  const size_t stride = (size_t)gridDim.x * blockDim.x;
  for (; i < n; i += stride) dst[i] = (bf16_t)src[i];
}

__global__ void zero_f32_kernel(float* __restrict__ dst, size_t n) {
  size_t i = (size_t)blockIdx.x * blockDim.x + threadIdx.x;
  const size_t stride = (size_t)gridDim.x * blockDim.x;
  for (; i < n; i += stride) dst[i] = 0.f;
}

// ---- Kernel 4: logits GEMM [4032 x 32000] = cat[4032x768] @ Wout^T + bout --
// Block tile 64M x 128N: each wave keeps 4 stacked 16x16 accumulators and
// reuses one B fragment across them -> 4x less L2 traffic on the 49MB
// L2-resident Wout (the binding resource at this arithmetic intensity).
// A tile (64x768 bf16, ~97KB LDS of the 320KB/WGP budget) is staged with the
// CDNA5 async global->LDS DMA and retired with s_wait_asynccnt.
__global__ void logits_kernel(const bf16_t* __restrict__ cat,
                              const bf16_t* __restrict__ Wb,
                              const float* __restrict__ bout,
                              float* __restrict__ dout) {
#define LPAD 776
  __shared__ alignas(16) bf16_t At[64][LPAD];  // 64 rows x 768 K (+8 pad)
  const int m0 = blockIdx.y * 64;
  const int tid = threadIdx.x;

  {  // async-DMA the A tile: 64 rows x 96 chunks of 16B, row-padded in LDS
    const bf16_t* gbase = cat + (size_t)m0 * CATD;
    for (int c = tid; c < 64 * 96; c += 256) {
      const int row = c / 96;
      const int off = (c - row * 96) * 8;     // bf16 elements (16B chunks)
      async_copy16(&At[row][off], gbase + (size_t)row * CATD + off);
    }
    async_wait0();
  }
  __syncthreads();

  const int wave = tid >> 5, lane = tid & 31;
  const int r = lane & 15, half = lane >> 4;
  const size_t n = (size_t)blockIdx.x * 128 + wave * 16 + r;
  const bf16_t* wrow = Wb + n * CATD;

  v8f acc[4];
#pragma unroll
  for (int ms = 0; ms < 4; ++ms)
    acc[ms] = (v8f){0.f, 0.f, 0.f, 0.f, 0.f, 0.f, 0.f, 0.f};

#pragma unroll
  for (int k0 = 0; k0 < CATD; k0 += 32) {
    __builtin_prefetch(wrow + k0 + 128, 0, 0);  // global_prefetch_b8 on weights
    v16bf b = frag_from_bf16row(wrow + k0 + half * 8);
#pragma unroll
    for (int ms = 0; ms < 4; ++ms) {
      v16bf a = frag_from_bf16row(&At[ms * 16 + r][k0 + half * 8]);
      acc[ms] = wmma_bf16(a, b, acc[ms]);
    }
  }

  const float bv = bout[n];
#pragma unroll
  for (int ms = 0; ms < 4; ++ms)
#pragma unroll
    for (int vv = 0; vv < 8; ++vv)
      dout[(size_t)(BATCH + m0 + ms * 16 + vv + half * 8) * VOCAB + n] =
          acc[ms][vv] + bv;
#undef LPAD
}

// ---- Kernel 5: per-row log-sum-exp + NLL (deterministic, one block/row) ----
__global__ void nll_kernel(const float* __restrict__ dout,
                           const int* __restrict__ y,
                           float* __restrict__ nll, float* __restrict__ msk) {
  __shared__ float red[256];
  const int r = blockIdx.x;
  const float* row = dout + (size_t)(BATCH + r) * VOCAB;
  const int tid = threadIdx.x;

  float m = -3.4e38f;
  for (int i = tid; i < VOCAB; i += 256) m = fmaxf(m, row[i]);
  red[tid] = m; __syncthreads();
  for (int s = 128; s > 0; s >>= 1) {
    if (tid < s) red[tid] = fmaxf(red[tid], red[tid + s]);
    __syncthreads();
  }
  m = red[0]; __syncthreads();

  float sum = 0.f;
  for (int i = tid; i < VOCAB; i += 256) sum += expf(row[i] - m);
  red[tid] = sum; __syncthreads();
  for (int s = 128; s > 0; s >>= 1) {
    if (tid < s) red[tid] += red[tid + s];
    __syncthreads();
  }
  if (tid == 0) {
    const float lse = m + logf(red[0]);
    const int tgt = y[r + BATCH];          // y[1:][t,b] == y[r + B]
    const float mv = (tgt != 0) ? 1.f : 0.f;
    nll[r] = (lse - row[tgt]) * mv;
    msk[r] = mv;
  }
}

// ---- Kernel 6: fixed-order loss reduction (bitwise deterministic) ----------
__global__ void loss_kernel(const float* __restrict__ nll,
                            const float* __restrict__ msk,
                            float* __restrict__ out_loss, int n) {
  __shared__ float a[256], b[256];
  const int tid = threadIdx.x;
  float sa = 0.f, sb = 0.f;
  for (int i = tid; i < n; i += 256) { sa += nll[i]; sb += msk[i]; }
  a[tid] = sa; b[tid] = sb; __syncthreads();
  for (int s = 128; s > 0; s >>= 1) {
    if (tid < s) { a[tid] += a[tid + s]; b[tid] += b[tid + s]; }
    __syncthreads();
  }
  if (tid == 0) out_loss[0] = a[0] / fmaxf(b[0], 1.f);
}

// ---- Host driver -----------------------------------------------------------
extern "C" void kernel_launch(void* const* d_in, const int* in_sizes, int n_in,
                              void* d_out, int out_size, void* d_ws, size_t ws_size,
                              hipStream_t stream) {
  (void)in_sizes; (void)n_in; (void)out_size; (void)ws_size;
  const int*   x      = (const int*)  d_in[0];
  const int*   xlens  = (const int*)  d_in[1];
  const int*   y      = (const int*)  d_in[2];
  const float* emb    = (const float*)d_in[3];
  const float* encWih = (const float*)d_in[4];
  const float* encWhh = (const float*)d_in[5];
  const float* encBih = (const float*)d_in[6];
  const float* encBhh = (const float*)d_in[7];
  const float* decWih = (const float*)d_in[8];
  const float* decWhh = (const float*)d_in[9];
  const float* decBih = (const float*)d_in[10];
  const float* decBhh = (const float*)d_in[11];
  const float* Wout   = (const float*)d_in[12];
  const float* bout   = (const float*)d_in[13];
  float* dout = (float*)d_out;

  char* ws = (char*)d_ws;
  auto alloc = [&](size_t bytes) -> char* {
    char* p = ws;
    ws += (bytes + 255) & ~(size_t)255;
    return p;
  };
  float*  xg   = (float*) alloc((size_t)T_STEPS * BATCH * G3 * 4);        // 25.2 MB
  float*  yg   = (float*) alloc((size_t)(T_STEPS - 1) * BATCH * G3 * 4);  // 24.8 MB
  float*  hg   = (float*) alloc((size_t)BATCH * G3 * 4);
  float*  hb   = (float*) alloc((size_t)BATCH * HDIM * 4);
  bf16_t* cat  = (bf16_t*)alloc((size_t)(T_STEPS - 1) * BATCH * CATD * 2);// 6.2 MB
  bf16_t* Wb   = (bf16_t*)alloc((size_t)VOCAB * CATD * 2);                // 49 MB (L2-resident)
  float*  nllv = (float*) alloc((size_t)(T_STEPS - 1) * BATCH * 4);
  float*  mskv = (float*) alloc((size_t)(T_STEPS - 1) * BATCH * 4);

  const int NROWS = (T_STEPS - 1) * BATCH;  // 4032

  // Independent prep
  f32_to_bf16_kernel<<<4096, 256, 0, stream>>>(Wout, Wb, (size_t)VOCAB * CATD);
  zero_f32_kernel<<<2048, 256, 0, stream>>>(dout, (size_t)BATCH * VOCAB);  // dec_out[0] = 0
  zero_f32_kernel<<<32, 256, 0, stream>>>(hb, (size_t)BATCH * HDIM);       // h0 = 0

  // Input-gate GEMMs (enc 256 row-tiles + dec 252 row-tiles = 508)
  gates_kernel<<<dim3(12, 508), 256, 0, stream>>>(x, y, emb, encWih, encBih,
                                                  decWih, decBih, xg, yg, cat);

  // Encoder scan (length-masked)
  for (int t = 0; t < T_STEPS; ++t) {
    gru_matmul_kernel<<<dim3(12, 4), 256, 0, stream>>>(hb, encWhh, encBhh, hg);
    gru_pointwise_kernel<<<128, 256, 0, stream>>>(xg + (size_t)t * BATCH * G3,
                                                  hg, hb, xlens, t, nullptr);
  }
  // Decoder scan (h0 = encoder final state, already in hb)
  for (int t = 0; t < T_STEPS - 1; ++t) {
    gru_matmul_kernel<<<dim3(12, 4), 256, 0, stream>>>(hb, decWhh, decBhh, hg);
    gru_pointwise_kernel<<<128, 256, 0, stream>>>(yg + (size_t)t * BATCH * G3,
                                                  hg, hb, nullptr, t,
                                                  cat + (size_t)t * BATCH * CATD);
  }

  // Output projection: 198 GFLOP bf16 WMMA (64M x 128N tiles, async-staged A)
  logits_kernel<<<dim3(VOCAB / 128, NROWS / 64), 256, 0, stream>>>(cat, Wb, bout, dout);

  // Loss
  nll_kernel<<<NROWS, 256, 0, stream>>>(dout, y, nllv, mskv);
  loss_kernel<<<1, 256, 0, stream>>>(nllv, mskv,
                                     dout + (size_t)T_STEPS * BATCH * VOCAB, NROWS);
}
